// FNO2d_54494545051959
// MI455X (gfx1250) — compile-verified, matrix operand order checked by reading, not run on hardware
//
#include <hip/hip_runtime.h>
#include <hip/hip_bf16.h>

typedef __attribute__((ext_vector_type(16))) _Float16 v16h;
typedef __attribute__((ext_vector_type(8)))  float    v8f;

#define PI_D 3.14159265358979323846

__device__ __forceinline__ float gelu_exact(float v) {
    return 0.5f * v * (1.0f + erff(v * 0.70710678118654752f));
}

// Load a 16-element f16 fragment (A or B side), k-range fully in-bounds.
// Pb already includes the lane-half offset (+8*sk); per-r offsets are wave-uniform.
__device__ __forceinline__ v16h frag_full(const float* __restrict__ Pb, long sk) {
    v16h f;
#pragma unroll
    for (int r = 0; r < 8; ++r) {
        long o = (long)(((r & 3) << 1) + ((r >> 2) << 4)) * sk;
        f[2 * r]     = (_Float16)Pb[o];
        f[2 * r + 1] = (_Float16)Pb[o + sk];
    }
    return f;
}

// K-tail fragment: unconditional in-bounds loads (address clamped back to the row
// start for k>=K) with value select-to-zero. No EXEC divergence.
__device__ __forceinline__ v16h frag_tail(const float* __restrict__ Pb, long sk,
                                          int kh, int K, long inv) {
    v16h f;
#pragma unroll
    for (int r = 0; r < 8; ++r) {
        int kp = ((r & 3) << 1) + ((r >> 2) << 4);
        bool v0 = (kh + kp)     < K;
        bool v1 = (kh + kp + 1) < K;
        long o = (long)kp * sk;
        float a0 = Pb[v0 ? o      : inv];
        float a1 = Pb[v1 ? o + sk : inv];
        f[2 * r]     = (_Float16)(v0 ? a0 : 0.f);
        f[2 * r + 1] = (_Float16)(v1 ? a1 : 0.f);
    }
    return f;
}

// ---------------------------------------------------------------------------
// Generic strided, batched WMMA GEMM:
//   C[m,n] = act( alpha * sum_k A[m,k]*B[k,n] + (Cin? Cin[m,n]:0) + (bias? bias[n]:0) )
// 128-thread blocks: 4 waves = 4 consecutive M-tiles; each wave does TN N-tiles,
// reusing its A fragment TN times. Batch z offsets: (z/bdiv)*outer + (z%bdiv)*inner.
// ---------------------------------------------------------------------------
template <int TN>
__global__ void wmma_gemm_f32(
    const float* __restrict__ A, const float* __restrict__ B, float* __restrict__ C,
    const float* __restrict__ Cin, const float* __restrict__ bias, float alpha,
    int M, int N, int K,
    long Asm, long Ask, long Bsk, long Bsn, long Csm, long Csn,
    int bdiv, long Ao, long Ai, long Bo, long Bi, long Cbo, long Cbi,
    int gelu)
{
    int z = blockIdx.z;
    int q = z / bdiv, rr = z - q * bdiv;
    A += (long)q * Ao + (long)rr * Ai;
    B += (long)q * Bo + (long)rr * Bi;
    long coff = (long)q * Cbo + (long)rr * Cbi;
    C += coff;
    if (Cin) Cin += coff;

    int wave = threadIdx.x >> 5;
    int lane = threadIdx.x & 31;
    int lq = lane & 15, half = lane >> 4;
    int m0 = (blockIdx.x * 4 + wave) * 16;
    int n0 = blockIdx.y * (16 * TN);

    // M/N edges: clamp only (an out-of-range row/col pollutes only unwritten outputs).
    int mrow = min(m0 + lq, M - 1);
    const float* Ab = A + (long)mrow * Asm + (long)(half << 3) * Ask;
    const float* Bb[TN];
#pragma unroll
    for (int t = 0; t < TN; ++t) {
        int nc = min(n0 + t * 16 + lq, N - 1);
        Bb[t] = B + (long)nc * Bsn + (long)(half << 3) * Bsk;
    }
    long invA = -(long)(half << 3) * Ask;
    long invB = -(long)(half << 3) * Bsk;

    v8f acc[TN];
#pragma unroll
    for (int t = 0; t < TN; ++t) acc[t] = {};

    int Kfull = K & ~31;
    int k0 = 0;
    for (; k0 < Kfull; k0 += 32) {
        if (k0 + 32 < K) __builtin_prefetch(Ab + 32 * Ask, 0, 1);
        v16h af = frag_full(Ab, Ask);
#pragma unroll
        for (int t = 0; t < TN; ++t) {
            v16h bf = frag_full(Bb[t], Bsk);
            acc[t] = __builtin_amdgcn_wmma_f32_16x16x32_f16(false, af, false, bf,
                                                            (short)0, acc[t], false, false);
        }
        Ab += 32 * Ask;
#pragma unroll
        for (int t = 0; t < TN; ++t) Bb[t] += 32 * Bsk;
    }
    if (k0 < K) {
        int kh = k0 + (half << 3);
        v16h af = frag_tail(Ab, Ask, kh, K, invA);
#pragma unroll
        for (int t = 0; t < TN; ++t) {
            v16h bf = frag_tail(Bb[t], Bsk, kh, K, invB);
            acc[t] = __builtin_amdgcn_wmma_f32_16x16x32_f16(false, af, false, bf,
                                                            (short)0, acc[t], false, false);
        }
    }

#pragma unroll
    for (int t = 0; t < TN; ++t) {
        int nc = n0 + t * 16 + lq;
        if (nc < N) {
            float bv = bias ? bias[nc] : 0.f;
#pragma unroll
            for (int r = 0; r < 8; ++r) {
                int m = m0 + r + (half << 3);
                if (m < M) {
                    long off = (long)m * Csm + (long)nc * Csn;
                    float v = alpha * acc[t][r] + bv + (Cin ? Cin[off] : 0.f);
                    if (gelu) v = gelu_exact(v);
                    C[off] = v;
                }
            }
        }
    }
}

// ---------------------------------------------------------------------------
// Fused head: out = fc2( gelu( fc1(x) + b1 ) ) + b2, per 16-pixel tile.
// X is NCHW [B,64,H,W]; hidden 16x256 tile staged in LDS as f16, both stages WMMA.
// ---------------------------------------------------------------------------
__global__ void mlp_head(const float* __restrict__ X,
                         const float* __restrict__ W1, const float* __restrict__ b1,
                         const float* __restrict__ W2, const float* __restrict__ b2,
                         float* __restrict__ out, int HW)
{
    __shared__ _Float16 hbuf[16 * 256];
    const int Cin = 64;
    int b = blockIdx.z;
    int p0 = blockIdx.x * 16;
    int lane = threadIdx.x & 31;
    int lq = lane & 15, half = lane >> 4;

    const float* Xb = X + (long)b * Cin * HW + (p0 + lq) + (long)(half << 3) * HW;
    v16h a0, a1;
#pragma unroll
    for (int r = 0; r < 8; ++r) {
        long kp = (long)(((r & 3) << 1) + ((r >> 2) << 4)) * HW;
        a0[2 * r]     = (_Float16)Xb[kp];
        a0[2 * r + 1] = (_Float16)Xb[kp + HW];
        a1[2 * r]     = (_Float16)Xb[kp + 32l * HW];
        a1[2 * r + 1] = (_Float16)Xb[kp + 33l * HW];
    }
    for (int nt = 0; nt < 16; ++nt) {
        int ncol = nt * 16 + lq;
        const float* Bc = W1 + (long)ncol * 64 + (half << 3);
        v8f acc = {};
        v16h bf;
#pragma unroll
        for (int r = 0; r < 8; ++r) {
            int k = ((r & 3) << 1) + ((r >> 2) << 4);
            bf[2 * r] = (_Float16)Bc[k]; bf[2 * r + 1] = (_Float16)Bc[k + 1];
        }
        acc = __builtin_amdgcn_wmma_f32_16x16x32_f16(false, a0, false, bf, (short)0, acc, false, false);
#pragma unroll
        for (int r = 0; r < 8; ++r) {
            int k = 32 + ((r & 3) << 1) + ((r >> 2) << 4);
            bf[2 * r] = (_Float16)Bc[k]; bf[2 * r + 1] = (_Float16)Bc[k + 1];
        }
        acc = __builtin_amdgcn_wmma_f32_16x16x32_f16(false, a1, false, bf, (short)0, acc, false, false);
        float bb = b1[ncol];
#pragma unroll
        for (int r = 0; r < 8; ++r) {
            int m = r + (half << 3);
            hbuf[m * 256 + ncol] = (_Float16)gelu_exact(acc[r] + bb);
        }
    }
    __syncthreads();

    bool nok = lq < 3;
    const float* Bc2 = W2 + (long)(nok ? lq : 0) * 256 + (half << 3);
    v8f acc2 = {};
    for (int s = 0; s < 8; ++s) {
        v16h af, bf;
#pragma unroll
        for (int r = 0; r < 8; ++r) {
            int kp = ((r & 3) << 1) + ((r >> 2) << 4);
            int k = s * 32 + kp + (half << 3);
            af[2 * r]     = hbuf[lq * 256 + k];
            af[2 * r + 1] = hbuf[lq * 256 + k + 1];
            float w0 = Bc2[s * 32 + kp];
            float w1 = Bc2[s * 32 + kp + 1];
            bf[2 * r]     = (_Float16)(nok ? w0 : 0.f);
            bf[2 * r + 1] = (_Float16)(nok ? w1 : 0.f);
        }
        acc2 = __builtin_amdgcn_wmma_f32_16x16x32_f16(false, af, false, bf, (short)0, acc2, false, false);
    }
    if (nok) {
        float bb = b2[lq];
#pragma unroll
        for (int r = 0; r < 8; ++r) {
            long p = p0 + r + (half << 3);
            out[((long)b * HW + p) * 3 + lq] = acc2[r] + bb;
        }
    }
}

// ----------------------------- elementwise kernels -------------------------
__global__ void k_pool2(const float* __restrict__ in, float* __restrict__ out,
                        int BC, int H, int W)
{
    int Wh = W >> 1, Hh = H >> 1;
    long total = (long)BC * Hh * Wh;
    long i = (long)blockIdx.x * blockDim.x + threadIdx.x;
    if (i >= total) return;
    int x = (int)(i % Wh); long t = i / Wh;
    int y = (int)(t % Hh); long bc = t / Hh;
    const float* p = in + ((long)bc * H + 2 * y) * W + 2 * x;
    out[i] = 0.25f * (p[0] + p[1] + p[W] + p[W + 1]);
}

__global__ void k_up2_into(const float* __restrict__ in, float* __restrict__ out,
                           int B, int Cs, int Ct, int Hs, int Ws)
{
    int Hd = 2 * Hs, Wd = 2 * Ws;
    long total = (long)B * Cs * Hd * Wd;
    long i = (long)blockIdx.x * blockDim.x + threadIdx.x;
    if (i >= total) return;
    int x = (int)(i % Wd); long t = i / Wd;
    int y = (int)(t % Hd); t /= Hd;
    int c = (int)(t % Cs); int b = (int)(t / Cs);
    float sy = (y + 0.5f) * 0.5f - 0.5f;
    float sx = (x + 0.5f) * 0.5f - 0.5f;
    int y0 = (int)floorf(sy); float wy = sy - (float)y0;
    int x0 = (int)floorf(sx); float wx = sx - (float)x0;
    int y0c = min(max(y0, 0), Hs - 1), y1c = min(max(y0 + 1, 0), Hs - 1);
    int x0c = min(max(x0, 0), Ws - 1), x1c = min(max(x0 + 1, 0), Ws - 1);
    const float* p = in + ((long)(b * Cs + c)) * Hs * Ws;
    float v = (1.f - wy) * ((1.f - wx) * p[y0c * Ws + x0c] + wx * p[y0c * Ws + x1c])
            +        wy  * ((1.f - wx) * p[y1c * Ws + x0c] + wx * p[y1c * Ws + x1c]);
    out[(((long)b * Ct + c) * Hd + y) * Wd + x] = v;
}

__global__ void k_copy_ch(const float* __restrict__ in, float* __restrict__ out,
                          int B, int Cs, int Ct, long HW, int coff)
{
    long total = (long)B * Cs * HW;
    long i = (long)blockIdx.x * blockDim.x + threadIdx.x;
    if (i >= total) return;
    long px = i % HW; long t = i / HW;
    int c = (int)(t % Cs); int b = (int)(t / Cs);
    out[((long)b * Ct + coff + c) * HW + px] = in[i];
}

// ------------------------------ twiddle kernels ----------------------------
__global__ void k_tw_fwd_x(float* C, float* S, int W) {
    int i = blockIdx.x * blockDim.x + threadIdx.x;
    if (i >= W * 16) return;
    int kx = i & 15, x = i >> 4;
    long ph = ((long)kx * x) % W;
    double a = -2.0 * PI_D * (double)ph / (double)W;  // e^{-i theta}
    C[i] = (float)cos(a);
    S[i] = (float)sin(a);
}
__global__ void k_tw_fwd_y(float* Cr, float* Ciy, int H, int m1) {
    int i = blockIdx.x * blockDim.x + threadIdx.x;
    if (i >= 32 * H) return;
    int y = i % H, r = i / H;
    if (r >= 2 * m1) { Cr[i] = 0.f; Ciy[i] = 0.f; return; }
    int ky = (r < m1) ? r : (H - 2 * m1 + r);
    long ph = ((long)ky * y) % H;
    double a = -2.0 * PI_D * (double)ph / (double)H;
    Cr[i] = (float)cos(a);
    Ciy[i] = (float)sin(a);
}
__global__ void k_tw_inv_y(float* Cc, float* Cs, int H, int m1) {
    int i = blockIdx.x * blockDim.x + threadIdx.x;
    if (i >= H * 32) return;
    int r = i & 31, y = i >> 5;
    if (r >= 2 * m1) { Cc[i] = 0.f; Cs[i] = 0.f; return; }
    int ky = (r < m1) ? r : (H - 2 * m1 + r);
    long ph = ((long)ky * y) % H;
    double a = 2.0 * PI_D * (double)ph / (double)H;
    Cc[i] = (float)(cos(a) / (double)H);
    Cs[i] = (float)(sin(a) / (double)H);
}
__global__ void k_tw_inv_x(float* C, float* S, int W, int m2) {
    int i = blockIdx.x * blockDim.x + threadIdx.x;
    if (i >= 16 * W) return;
    int x = i % W, kx = i / W;
    if (kx >= m2) { C[i] = 0.f; S[i] = 0.f; return; }
    double cc = (kx == 0) ? 1.0 : 2.0;   // rfft Hermitian doubling
    long ph = ((long)kx * x) % W;
    double a = 2.0 * PI_D * (double)ph / (double)W;
    C[i] = (float)(cc * cos(a) / (double)W);
    S[i] = (float)(cc * sin(a) / (double)W);
}

// ------------------------------- host helpers ------------------------------
struct Tw { float *cfx, *sfx, *wyr, *wyi, *wic, *wis, *cix, *six; int H, m1; };

static void init_tw(hipStream_t s, const Tw& t) {
    int H = t.H;
    k_tw_fwd_x<<<(H * 16 + 255) / 256, 256, 0, s>>>(t.cfx, t.sfx, H);
    k_tw_fwd_y<<<(32 * H + 255) / 256, 256, 0, s>>>(t.wyr, t.wyi, H, t.m1);
    k_tw_inv_y<<<(H * 32 + 255) / 256, 256, 0, s>>>(t.wic, t.wis, H, t.m1);
    k_tw_inv_x<<<(16 * H + 255) / 256, 256, 0, s>>>(t.cix, t.six, H, t.m1);
}

static void gemm(hipStream_t s, const float* A, const float* B, float* C,
                 const float* Cin, const float* bias, float alpha,
                 int M, int N, int K,
                 long Asm, long Ask, long Bsk, long Bsn, long Csm, long Csn,
                 int nb, int bdiv, long Ao, long Ai, long Bo, long Bi,
                 long Cbo, long Cbi, int gelu)
{
    if (N >= 32) {
        dim3 grid((M + 63) / 64, (N + 31) / 32, nb);
        wmma_gemm_f32<2><<<grid, 128, 0, s>>>(A, B, C, Cin, bias, alpha, M, N, K,
                                              Asm, Ask, Bsk, Bsn, Csm, Csn,
                                              bdiv, Ao, Ai, Bo, Bi, Cbo, Cbi, gelu);
    } else {
        dim3 grid((M + 63) / 64, (N + 15) / 16, nb);
        wmma_gemm_f32<1><<<grid, 128, 0, s>>>(A, B, C, Cin, bias, alpha, M, N, K,
                                              Asm, Ask, Bsk, Bsn, Csm, Csn,
                                              bdiv, Ao, Ai, Bo, Bi, Cbo, Cbi, gelu);
    }
}

// Full spectral conv + fused 1x1 conv + GELU: outC = gelu(spectral(act) + conv(act) + b)
static void spectral_block(hipStream_t s, const float* act, float* outC,
                           int Ci, int Co, int H, int W, int m1,
                           const float* w1, const float* w2, const Tw& tw,
                           float* XR, float* XI, float* FR, float* FI,
                           float* ORr, float* OIr, float* TR, float* TI,
                           const float* convw, const float* convb)
{
    const int B = 4, m2 = m1;
    long HWl = (long)H * W;
    // 1) truncated DFT along x (real input -> XR, XI), rows = (b,ci,y)
    int M1 = B * Ci * H;
    gemm(s, act, tw.cfx, XR, nullptr, nullptr, 1.f, M1, 16, W, W, 1, 16, 1, 16, 1,
         1, 1, 0, 0, 0, 0, 0, 0, 0);
    gemm(s, act, tw.sfx, XI, nullptr, nullptr, 1.f, M1, 16, W, W, 1, 16, 1, 16, 1,
         1, 1, 0, 0, 0, 0, 0, 0, 0);
    // 2) truncated DFT along y, batched per (b,ci): F = Wy x X (complex)
    int nb = B * Ci;
    long Bb = (long)H * 16, Cb = 512;
    gemm(s, tw.wyr, XR, FR, nullptr, nullptr,  1.f, 32, 16, H, H, 1, 16, 1, 16, 1, nb, 1, 0, 0, Bb, 0, Cb, 0, 0);
    gemm(s, tw.wyi, XI, FR, FR,     nullptr, -1.f, 32, 16, H, H, 1, 16, 1, 16, 1, nb, 1, 0, 0, Bb, 0, Cb, 0, 0);
    gemm(s, tw.wyr, XI, FI, nullptr, nullptr,  1.f, 32, 16, H, H, 1, 16, 1, 16, 1, nb, 1, 0, 0, Bb, 0, Cb, 0, 0);
    gemm(s, tw.wyi, XR, FI, FI,     nullptr,  1.f, 32, 16, H, H, 1, 16, 1, 16, 1, nb, 1, 0, 0, Bb, 0, Cb, 0, 0);
    // 3) per-mode channel mixing: o[b,co] = sum_ci F[b,ci] * w[ci,co] (complex), batch = mode
    long AsmM = (long)Ci * 512, AskM = 512;
    long BskM = (long)Co * m1 * m2 * 2, BsnM = (long)m1 * m2 * 2;
    long CsmM = (long)Co * 512, CsnM = 512;
    int nbm = m1 * m2;
    for (int hs = 0; hs < 2; ++hs) {
        const float* w = hs ? w2 : w1;
        long fo = hs ? (long)m1 * 16 : 0;
        const float* fr = FR + fo; const float* fi = FI + fo;
        float* orp = ORr + fo;     float* oip = OIr + fo;
        gemm(s, fr, w,     orp, nullptr, nullptr,  1.f, B, Co, Ci, AsmM, AskM, BskM, BsnM, CsmM, CsnM,
             nbm, m2, 16, 1, (long)m2 * 2, 2, 16, 1, 0);
        gemm(s, fi, w + 1, orp, orp,     nullptr, -1.f, B, Co, Ci, AsmM, AskM, BskM, BsnM, CsmM, CsnM,
             nbm, m2, 16, 1, (long)m2 * 2, 2, 16, 1, 0);
        gemm(s, fr, w + 1, oip, nullptr, nullptr,  1.f, B, Co, Ci, AsmM, AskM, BskM, BsnM, CsmM, CsnM,
             nbm, m2, 16, 1, (long)m2 * 2, 2, 16, 1, 0);
        gemm(s, fi, w,     oip, oip,     nullptr,  1.f, B, Co, Ci, AsmM, AskM, BskM, BsnM, CsmM, CsnM,
             nbm, m2, 16, 1, (long)m2 * 2, 2, 16, 1, 0);
    }
    // 4) inverse DFT along y (1/H folded into twiddle), batched per (b,co)
    int nbc = B * Co;
    long Tb = (long)H * 16;
    gemm(s, tw.wic, ORr, TR, nullptr, nullptr,  1.f, H, 16, 32, 32, 1, 16, 1, 16, 1, nbc, 1, 0, 0, 512, 0, Tb, 0, 0);
    gemm(s, tw.wis, OIr, TR, TR,     nullptr, -1.f, H, 16, 32, 32, 1, 16, 1, 16, 1, nbc, 1, 0, 0, 512, 0, Tb, 0, 0);
    gemm(s, tw.wis, ORr, TI, nullptr, nullptr,  1.f, H, 16, 32, 32, 1, 16, 1, 16, 1, nbc, 1, 0, 0, 512, 0, Tb, 0, 0);
    gemm(s, tw.wic, OIr, TI, TI,     nullptr,  1.f, H, 16, 32, 32, 1, 16, 1, 16, 1, nbc, 1, 0, 0, 512, 0, Tb, 0, 0);
    // 5) inverse rfft along x (Hermitian doubling in twiddle): real output
    int M5 = B * Co * H;
    gemm(s, TR, tw.cix, outC, nullptr, nullptr,  1.f, M5, W, 16, 16, 1, W, 1, W, 1, 1, 1, 0, 0, 0, 0, 0, 0, 0);
    gemm(s, TI, tw.six, outC, outC,   nullptr, -1.f, M5, W, 16, 16, 1, W, 1, W, 1, 1, 1, 0, 0, 0, 0, 0, 0, 0);
    // 6) fused 1x1 conv + bias + GELU (NCHW), batched over b
    gemm(s, act, convw, outC, outC, convb, 1.f, (int)HWl, Co, Ci,
         1, HWl, 1, Ci, 1, HWl,
         B, 1, (long)Ci * HWl, 0, 0, 0, (long)Co * HWl, 0, 1);
}

// ---------------------------------------------------------------------------
extern "C" void kernel_launch(void* const* d_in, const int* in_sizes, int n_in,
                              void* d_out, int out_size, void* d_ws, size_t ws_size,
                              hipStream_t stream)
{
    (void)in_sizes; (void)n_in; (void)out_size; (void)ws_size;
    const float* x_in   = (const float*)d_in[0];
    const float* fcin_w = (const float*)d_in[1];
    const float* fcin_b = (const float*)d_in[2];
    const float* sc1_w1 = (const float*)d_in[3];
    const float* sc1_w2 = (const float*)d_in[4];
    const float* c1_w   = (const float*)d_in[5];
    const float* c1_b   = (const float*)d_in[6];
    const float* sc2_w1 = (const float*)d_in[7];
    const float* sc2_w2 = (const float*)d_in[8];
    const float* c2_w   = (const float*)d_in[9];
    const float* c2_b   = (const float*)d_in[10];
    const float* scb_w1 = (const float*)d_in[11];
    const float* scb_w2 = (const float*)d_in[12];
    const float* cbw    = (const float*)d_in[13];
    const float* cbb    = (const float*)d_in[14];
    const float* su2_w1 = (const float*)d_in[15];
    const float* su2_w2 = (const float*)d_in[16];
    const float* u2_w   = (const float*)d_in[17];
    const float* u2_b   = (const float*)d_in[18];
    const float* su1_w1 = (const float*)d_in[19];
    const float* su1_w2 = (const float*)d_in[20];
    const float* u1_w   = (const float*)d_in[21];
    const float* u1_b   = (const float*)d_in[22];
    const float* fc1_w  = (const float*)d_in[23];
    const float* fc1_b  = (const float*)d_in[24];
    const float* fc2_w  = (const float*)d_in[25];
    const float* fc2_b  = (const float*)d_in[26];

    const long HW1 = 65536, HW2 = 16384, HW3 = 4096;

    float* W = (float*)d_ws;
    size_t off = 0;
    auto alloc = [&](size_t n) { float* p = W + off; off += n; return p; };

    float* v    = alloc(16777216);   // [4,64,256,256]
    float* x1   = alloc(16777216);
    float* x1d  = alloc(4194304);
    float* x2   = alloc(6291456);
    float* x2d  = alloc(1572864);
    float* xb   = alloc(2097152);
    float* x2c  = alloc(14680064);   // [4,224,128,128]
    float* x2o  = alloc(6291456);
    float* x1c  = alloc(41943040);   // [4,160,256,256]
    float* XR   = alloc(2621440);
    float* XI   = alloc(2621440);
    float* FR   = alloc(458752);
    float* FI   = alloc(458752);
    float* ORr  = alloc(262144);
    float* OIr  = alloc(262144);
    float* TR   = alloc(1048576);
    float* TI   = alloc(1048576);

    Tw t256 = { alloc(256*16), alloc(256*16), alloc(32*256), alloc(32*256),
                alloc(256*32), alloc(256*32), alloc(16*256), alloc(16*256), 256, 12 };
    Tw t128 = { alloc(128*16), alloc(128*16), alloc(32*128), alloc(32*128),
                alloc(128*32), alloc(128*32), alloc(16*128), alloc(16*128), 128, 8 };
    Tw t64  = { alloc(64*16),  alloc(64*16),  alloc(32*64),  alloc(32*64),
                alloc(64*32),  alloc(64*32),  alloc(16*64),  alloc(16*64),  64, 4 };
    float* x1o = v;  // v is dead by the time x1o is produced

    init_tw(stream, t256);
    init_tw(stream, t128);
    init_tw(stream, t64);

    // lift: v[b,co,p] = x[b,p,ci] @ fcin_w^T + b   (NHWC -> NCHW)
    gemm(stream, x_in, fcin_w, v, nullptr, fcin_b, 1.f,
         (int)HW1, 64, 6, 6, 1, 1, 6, 1, HW1,
         4, 1, HW1 * 6, 0, 0, 0, 64 * HW1, 0, 0);

    // level 1
    spectral_block(stream, v, x1, 64, 64, 256, 256, 12, sc1_w1, sc1_w2, t256,
                   XR, XI, FR, FI, ORr, OIr, TR, TI, c1_w, c1_b);
    k_pool2<<<(4 * 64 * HW2 + 255) / 256, 256, 0, stream>>>(x1, x1d, 4 * 64, 256, 256);

    // level 2
    spectral_block(stream, x1d, x2, 64, 96, 128, 128, 8, sc2_w1, sc2_w2, t128,
                   XR, XI, FR, FI, ORr, OIr, TR, TI, c2_w, c2_b);
    k_pool2<<<(4 * 96 * HW3 + 255) / 256, 256, 0, stream>>>(x2, x2d, 4 * 96, 128, 128);

    // bottom
    spectral_block(stream, x2d, xb, 96, 128, 64, 64, 4, scb_w1, scb_w2, t64,
                   XR, XI, FR, FI, ORr, OIr, TR, TI, cbw, cbb);

    // x2c = concat(up(xb), x2)
    k_up2_into<<<(4 * 128 * HW2 + 255) / 256, 256, 0, stream>>>(xb, x2c, 4, 128, 224, 64, 64);
    k_copy_ch<<<(4 * 96 * HW2 + 255) / 256, 256, 0, stream>>>(x2, x2c, 4, 96, 224, HW2, 128);

    // up-level 2
    spectral_block(stream, x2c, x2o, 224, 96, 128, 128, 8, su2_w1, su2_w2, t128,
                   XR, XI, FR, FI, ORr, OIr, TR, TI, u2_w, u2_b);

    // x1c = concat(up(x2o), x1)
    k_up2_into<<<(4 * 96 * HW1 + 255) / 256, 256, 0, stream>>>(x2o, x1c, 4, 96, 160, 128, 128);
    k_copy_ch<<<(4 * 64 * HW1 + 255) / 256, 256, 0, stream>>>(x1, x1c, 4, 64, 160, HW1, 96);

    // up-level 1
    spectral_block(stream, x1c, x1o, 160, 64, 256, 256, 12, su1_w1, su1_w2, t256,
                   XR, XI, FR, FI, ORr, OIr, TR, TI, u1_w, u1_b);

    // fused fc1 -> GELU -> fc2 head (WMMA, LDS-staged hidden tile)
    mlp_head<<<dim3(4096, 1, 4), 32, 0, stream>>>(x1o, fc1_w, fc1_b, fc2_w, fc2_b,
                                                  (float*)d_out, (int)HW1);
}
